// BiLSTM_CRF_MODIFY_PARALLEL_89850715833154
// MI455X (gfx1250) — compile-verified, matrix operand order checked by reading, not run on hardware
//
#include <hip/hip_runtime.h>
#include <hip/hip_bf16.h>
#include <math.h>

#define T_LEN 256
#define BATCH 128
#define EDIM  256
#define H2    256
#define G4    1024   // 4*H2
#define HID   512
#define NT    10
#define NEGV  (-10000.0f)
#define START_TAG 8
#define STOP_TAG  9

typedef __bf16 bf16t;
typedef bf16t v16bf __attribute__((ext_vector_type(16)));
typedef float v8f   __attribute__((ext_vector_type(8)));

struct U128 { unsigned int x, y, z, w; };
union Frag { v16bf v; U128 q[2]; };

__device__ __forceinline__ unsigned short f2bf(float f) {
  unsigned int u = __float_as_uint(f);
  u += 0x7fffu + ((u >> 16) & 1u);          // round-to-nearest-even
  return (unsigned short)(u >> 16);
}

__device__ __forceinline__ v8f v8f_zero() {
  v8f z;
#pragma unroll
  for (int i = 0; i < 8; ++i) z[i] = 0.0f;
  return z;
}

// ---------------- fp32 -> bf16 convert ----------------
__global__ void cvt_bf16_kernel(const float* __restrict__ src,
                                unsigned short* __restrict__ dst, int n) {
  int i = blockIdx.x * blockDim.x + threadIdx.x;
  if (i < n) dst[i] = f2bf(src[i]);
}

// ---------------- embedding gather -> bf16 [T*B, E] ----------------
__global__ void gather_embed_kernel(const int* __restrict__ sent,
                                    const float* __restrict__ embed,
                                    unsigned short* __restrict__ xbf) {
  int i = blockIdx.x * blockDim.x + threadIdx.x;   // 0 .. T*B*E-1
  int e   = i & (EDIM - 1);
  int row = i >> 8;            // t*B + b
  int t   = row >> 7;          // /128
  int b   = row & 127;
  int tok = sent[b * T_LEN + t];
  xbf[i]  = f2bf(embed[(size_t)tok * EDIM + e]);
}

// ---------------- input-projection GEMM (WMMA bf16) ----------------
// Xproj[dir][row, col] = x[row,:] @ Wih_dir[col,:] + bih[col] + bhh[col]
__global__ __launch_bounds__(512) void xproj_gemm_kernel(
    const unsigned short* __restrict__ xbf,     // [32768,256] bf16
    const unsigned short* __restrict__ wih_bf,  // [2][1024,256] bf16
    const float* __restrict__ bih_f, const float* __restrict__ bhh_f,
    const float* __restrict__ bih_b, const float* __restrict__ bhh_b,
    float* __restrict__ xproj)                  // [2][32768,1024]
{
  unsigned tid  = threadIdx.x;
  unsigned wave = tid >> 5, lane = tid & 31u;
  unsigned g = lane >> 4, nl = lane & 15u;

  unsigned id  = blockIdx.x * 16u + wave;       // tile id 0..262143
  unsigned dir = id >> 17;                      // 131072 tiles per dir
  unsigned rem = id & 131071u;
  unsigned mt  = rem >> 6, nt = rem & 63u;

  const unsigned short* wih = wih_bf + (size_t)dir * G4 * EDIM;

  v8f acc = v8f_zero();
  unsigned arow = mt * 16u + nl;
  unsigned brow = nt * 16u + nl;
#pragma unroll
  for (int kk = 0; kk < 8; ++kk) {
    unsigned kb1 = (unsigned)kk * 32u + g * 8u;
    Frag a, b;
    a.q[0] = *(const U128*)(xbf + (size_t)arow * EDIM + kb1);
    a.q[1] = *(const U128*)(xbf + (size_t)arow * EDIM + kb1 + 16);
    b.q[0] = *(const U128*)(wih + (size_t)brow * EDIM + kb1);
    b.q[1] = *(const U128*)(wih + (size_t)brow * EDIM + kb1 + 16);
    acc = __builtin_amdgcn_wmma_f32_16x16x32_bf16(false, a.v, false, b.v,
                                                  (short)0, acc, false, false);
  }
  unsigned col = nt * 16u + nl;
  float bias = dir ? (bih_b[col] + bhh_b[col]) : (bih_f[col] + bhh_f[col]);
  float* out = xproj + (size_t)dir * 32768u * 1024u;
#pragma unroll
  for (int r = 0; r < 8; ++r) {
    unsigned row = mt * 16u + (unsigned)r + 8u * g;
    out[(size_t)row * G4 + col] = acc[r] + bias;
  }
}

// ---------------- recurrent LSTM (WMMA, Whh resident in VGPRs) ----------------
// grid (8 batch-tiles, 2 dirs), 512 threads = 16 waves; wave owns hidden tile jn.
// Double-buffered h in LDS: one barrier per timestep. Xproj (incl. biases) is
// loaded straight into the WMMA accumulators before the barrier so the global
// load latency overlaps the barrier wait + WMMA chain.
__global__ __launch_bounds__(512, 1) void lstm_rec_kernel(
    const unsigned short* __restrict__ whh_bf,  // [2][1024,256] bf16
    const float* __restrict__ xproj,            // [2][32768,1024]
    float* __restrict__ hout)                   // [T*B, 512]
{
  __shared__ __align__(16) unsigned short h_lds[2][16][256];

  unsigned tid  = threadIdx.x;
  unsigned wave = tid >> 5, lane = tid & 31u;
  unsigned g = lane >> 4, nl = lane & 15u;
  unsigned jn  = wave;            // 0..15 hidden tile
  unsigned bm  = blockIdx.x;      // 0..7 batch tile
  unsigned dir = blockIdx.y;      // 0..1

  const unsigned short* whh = whh_bf + (size_t)dir * G4 * H2;
  const float* xp = xproj + (size_t)dir * 32768u * 1024u;

  // Preload all 32 Whh B-fragments (4 gates x 8 k-steps) into registers
  Frag bfr[4][8];
#pragma unroll
  for (int gate = 0; gate < 4; ++gate) {
    unsigned brow = (unsigned)gate * 256u + jn * 16u + nl;
#pragma unroll
    for (int kk = 0; kk < 8; ++kk) {
      unsigned kb1 = (unsigned)kk * 32u + g * 8u;
      bfr[gate][kk].q[0] = *(const U128*)(whh + (size_t)brow * H2 + kb1);
      bfr[gate][kk].q[1] = *(const U128*)(whh + (size_t)brow * H2 + kb1 + 16);
    }
  }

  // zero h state (buffer 0 is read at s=0)
  for (unsigned i = tid; i < 2u * 16u * 256u; i += 512u)
    ((unsigned short*)h_lds)[i] = 0;

  v8f c = v8f_zero();
  unsigned colb = jn * 16u + nl;

  for (int s = 0; s < T_LEN; ++s) {
    int tsrc = dir ? (T_LEN - 1 - s) : s;
    unsigned cur = (unsigned)s & 1u, nxt = cur ^ 1u;

    // Seed accumulators with Xproj (= x@Wih.T + biases); independent of h,
    // so these global loads issue before the barrier.
    v8f acc0, acc1, acc2, acc3;
#pragma unroll
    for (int r = 0; r < 8; ++r) {
      unsigned brows = bm * 16u + (unsigned)r + 8u * g;
      size_t rowoff = ((size_t)tsrc * BATCH + brows) * G4;
      acc0[r] = xp[rowoff + 0 * 256 + colb];
      acc1[r] = xp[rowoff + 1 * 256 + colb];
      acc2[r] = xp[rowoff + 2 * 256 + colb];
      acc3[r] = xp[rowoff + 3 * 256 + colb];
    }

    __syncthreads();              // h_lds[cur] from previous step visible

#pragma unroll
    for (int kk = 0; kk < 8; ++kk) {
      unsigned kb1 = (unsigned)kk * 32u + g * 8u;
      Frag a;
      a.q[0] = *(const U128*)&h_lds[cur][nl][kb1];
      a.q[1] = *(const U128*)&h_lds[cur][nl][kb1 + 16];
      acc0 = __builtin_amdgcn_wmma_f32_16x16x32_bf16(false, a.v, false, bfr[0][kk].v, (short)0, acc0, false, false);
      acc1 = __builtin_amdgcn_wmma_f32_16x16x32_bf16(false, a.v, false, bfr[1][kk].v, (short)0, acc1, false, false);
      acc2 = __builtin_amdgcn_wmma_f32_16x16x32_bf16(false, a.v, false, bfr[2][kk].v, (short)0, acc2, false, false);
      acc3 = __builtin_amdgcn_wmma_f32_16x16x32_bf16(false, a.v, false, bfr[3][kk].v, (short)0, acc3, false, false);
    }

#pragma unroll
    for (int r = 0; r < 8; ++r) {
      unsigned brows = bm * 16u + (unsigned)r + 8u * g;
      float si = 1.0f / (1.0f + __expf(-acc0[r]));
      float sf = 1.0f / (1.0f + __expf(-acc1[r]));
      float so = 1.0f / (1.0f + __expf(-acc3[r]));
      float cn = sf * c[r] + si * tanhf(acc2[r]);
      c[r] = cn;
      float h = so * tanhf(cn);
      hout[((size_t)tsrc * BATCH + brows) * HID + dir * H2 + colb] = h;
      h_lds[nxt][(unsigned)r + 8u * g][colb] = f2bf(h);
    }
  }
}

// ---------------- feats = h @ W_h2t.T + b ----------------
__global__ void feats_kernel(const float* __restrict__ hout,  // [T*B, 512]
                             const float* __restrict__ w,     // [10, 512]
                             const float* __restrict__ b,
                             float* __restrict__ feats) {     // [T*B, 10]
  int idx = blockIdx.x * blockDim.x + threadIdx.x;            // 32768*16
  int row = idx >> 4, n = idx & 15;
  if (n >= NT) return;
  const float* hr = hout + (size_t)row * HID;
  const float* wr = w + (size_t)n * HID;
  float s = b[n];
  for (int k = 0; k < HID; ++k) s += hr[k] * wr[k];
  feats[(size_t)row * NT + n] = s;
}

// ---------------- CRF forward + gold score + reduce ----------------
__global__ void crf_kernel(const float* __restrict__ feats,   // [T*B, 10]
                           const int* __restrict__ tags,      // [B, T]
                           const float* __restrict__ trans,   // [10,10]
                           float* __restrict__ out) {
  __shared__ float s_tr[NT * NT];
  __shared__ float red[BATCH];
  int b = threadIdx.x;
  if (b < NT * NT) s_tr[b] = trans[b];
  __syncthreads();

  float alpha[NT];
#pragma unroll
  for (int j = 0; j < NT; ++j) alpha[j] = NEGV;
  alpha[START_TAG] = 0.0f;

  float gold = 0.0f;
  int prev = START_TAG;
  for (int t = 0; t < T_LEN; ++t) {
    const float* ft = feats + ((size_t)t * BATCH + b) * NT;
    float f[NT];
#pragma unroll
    for (int j = 0; j < NT; ++j) f[j] = ft[j];
    float na[NT];
#pragma unroll
    for (int nx = 0; nx < NT; ++nx) {
      float m = -1e30f;
#pragma unroll
      for (int p = 0; p < NT; ++p) m = fmaxf(m, alpha[p] + s_tr[nx * NT + p]);
      float ssum = 0.0f;
#pragma unroll
      for (int p = 0; p < NT; ++p) ssum += __expf(alpha[p] + s_tr[nx * NT + p] - m);
      na[nx] = m + __logf(ssum) + f[nx];
    }
#pragma unroll
    for (int j = 0; j < NT; ++j) alpha[j] = na[j];
    int tg = tags[b * T_LEN + t];
    gold += s_tr[tg * NT + prev] + f[tg];
    prev = tg;
  }
  gold += s_tr[STOP_TAG * NT + prev];

  float m = -1e30f;
#pragma unroll
  for (int j = 0; j < NT; ++j) m = fmaxf(m, alpha[j] + s_tr[STOP_TAG * NT + j]);
  float ssum = 0.0f;
#pragma unroll
  for (int j = 0; j < NT; ++j) ssum += __expf(alpha[j] + s_tr[STOP_TAG * NT + j] - m);
  float fwd = m + __logf(ssum);

  red[b] = fwd - gold;
  __syncthreads();
  for (int off = 64; off > 0; off >>= 1) {
    if (b < off) red[b] += red[b + off];
    __syncthreads();
  }
  if (b == 0) out[0] = red[0];
}

// ---------------- host launcher ----------------
extern "C" void kernel_launch(void* const* d_in, const int* in_sizes, int n_in,
                              void* d_out, int out_size, void* d_ws, size_t ws_size,
                              hipStream_t stream) {
  const int*   sent  = (const int*)d_in[0];
  const int*   tags  = (const int*)d_in[1];
  const float* embed = (const float*)d_in[2];
  const float* Wih_f = (const float*)d_in[3];
  const float* Whh_f = (const float*)d_in[4];
  const float* bih_f = (const float*)d_in[5];
  const float* bhh_f = (const float*)d_in[6];
  const float* Wih_b = (const float*)d_in[7];
  const float* Whh_b = (const float*)d_in[8];
  const float* bih_b = (const float*)d_in[9];
  const float* bhh_b = (const float*)d_in[10];
  const float* W_h2t = (const float*)d_in[11];
  const float* b_h2t = (const float*)d_in[12];
  const float* trans = (const float*)d_in[13];

  char* ws = (char*)d_ws;
  unsigned short* xbf    = (unsigned short*)(ws);                 // 16,777,216 B
  unsigned short* wih_bf = (unsigned short*)(ws + 16777216);      //  1,048,576 B
  unsigned short* whh_bf = (unsigned short*)(ws + 17825792);      //  1,048,576 B
  float* xproj           = (float*)(ws + 18874368);               // 268,435,456 B
  float* hout            = (float*)(ws + 287309824);              //  67,108,864 B
  float* feats           = (float*)(ws + 354418688);              //   1,310,720 B

  const int nW = G4 * EDIM;  // 262144
  cvt_bf16_kernel<<<(nW + 255) / 256, 256, 0, stream>>>(Wih_f, wih_bf, nW);
  cvt_bf16_kernel<<<(nW + 255) / 256, 256, 0, stream>>>(Wih_b, wih_bf + nW, nW);
  cvt_bf16_kernel<<<(nW + 255) / 256, 256, 0, stream>>>(Whh_f, whh_bf, nW);
  cvt_bf16_kernel<<<(nW + 255) / 256, 256, 0, stream>>>(Whh_b, whh_bf + nW, nW);

  gather_embed_kernel<<<(T_LEN * BATCH * EDIM) / 256, 256, 0, stream>>>(sent, embed, xbf);

  xproj_gemm_kernel<<<16384, 512, 0, stream>>>(xbf, wih_bf, bih_f, bhh_f,
                                               bih_b, bhh_b, xproj);

  lstm_rec_kernel<<<dim3(8, 2), 512, 0, stream>>>(whh_bf, xproj, hout);

  feats_kernel<<<(32768 * 16) / 256, 256, 0, stream>>>(hout, W_h2t, b_h2t, feats);

  crf_kernel<<<1, BATCH, 0, stream>>>(feats, tags, trans, (float*)d_out);
}
